// DecoderRNN_60035052863763
// MI455X (gfx1250) — matrix operand order repaired and checked by reading, not compiled, not run on hardware
//
#include <hip/hip_runtime.h>
#include <hip/hip_bf16.h>

// CDNA5 / gfx1250: wave32, WMMA bf16 path.
typedef __attribute__((ext_vector_type(16))) __bf16 v16bf;
typedef __attribute__((ext_vector_type(8)))  __bf16 v8bf;
typedef __attribute__((ext_vector_type(8)))  float  v8f;

#define BT     32            // batch rows per workgroup
#define AS     200           // activation row stride (192 + 8 pad), bf16 elems
#define WS     200           // weight row stride (192 + 8 pad), bf16 elems
#define NT     256           // threads per block (8 wave32)
#define PSTEPS 64
#define ZELT   (BT * 63)     // z elements staged per step (2016)

__device__ __forceinline__ float sigmoidf_(float v) {
  return 1.0f / (1.0f + __expf(-v));
}

extern "C" __global__ __launch_bounds__(NT, 1)
void lstm_decoder_wmma(const float* __restrict__ x,
                       const float* __restrict__ z,
                       const float* __restrict__ h0,
                       const float* __restrict__ c0,
                       const float* __restrict__ W_ih,
                       const float* __restrict__ W_hh,
                       const float* __restrict__ b_ih,
                       const float* __restrict__ b_hh,
                       const float* __restrict__ W_out,
                       const float* __restrict__ b_out,
                       float* __restrict__ out)
{
  extern __shared__ char smem[];
  __bf16* sW    = (__bf16*)smem;             // 512 x WS bf16 (W_ih | W_hh), N-major
  __bf16* sAct  = sW + 512 * WS;             // BT x AS bf16: [out | z_t | h]
  float*  sWout = (float*)(sAct + BT * AS);  // 128 f32

  const int tid  = threadIdx.x;
  const int lane = tid & 31;
  const int wave = tid >> 5;
  const int b0   = blockIdx.x * BT;

  // ---------------- one-time setup ----------------
  // Fused weight matrix: row g (0..511) = [W_ih[g, 0:64] | W_hh[g, 0:128]] in bf16.
  for (int idx = tid; idx < 512 * 192; idx += NT) {
    int g = idx / 192, k = idx % 192;
    float v = (k < 64) ? W_ih[g * 64 + k] : W_hh[g * 128 + (k - 64)];
    sW[g * WS + k] = (__bf16)v;
  }
  if (tid < 128) sWout[tid] = W_out[tid];
  // h0 into activation cols 64..191; x[:, -1] into col 0 (initial "out").
  for (int idx = tid; idx < BT * 128; idx += NT) {
    int r = idx >> 7, j = idx & 127;
    sAct[r * AS + 64 + j] = (__bf16)h0[(size_t)(b0 + r) * 128 + j];
  }
  if (tid < BT)
    sAct[tid * AS + 0] = (__bf16)x[(size_t)(b0 + tid) * 128 + 127];

  // Per-wave tile ownership: rows [mtile*16, +16), hidden col tiles {hbase, hbase+4}.
  const int mtile = wave & 1;
  const int hbase = wave >> 1;
  const int n     = lane & 15;   // N / M-in-lane index
  const int half  = lane >> 4;   // lane half selects K / M sub-block

  // Cell state c in registers (C-fragment layout) for all 64 steps.
  v8f   creg[2];
  float biasr[2][4];
  #pragma unroll
  for (int p = 0; p < 2; ++p) {
    const int hcol = (hbase + 4 * p) * 16 + n;
    #pragma unroll
    for (int v = 0; v < 8; ++v) {
      const int m = v + 8 * half;
      creg[p][v] = c0[(size_t)(b0 + mtile * 16 + m) * 128 + hcol];
    }
    #pragma unroll
    for (int gi = 0; gi < 4; ++gi)
      biasr[p][gi] = b_ih[gi * 128 + hcol] + b_hh[gi * 128 + hcol];
  }
  const float boutv = b_out[0];

  // z register pipeline: 8 slots per thread, index CLAMPED (no predication:
  // the ~32 tail threads redundantly handle element 2015 with identical
  // values -> benign same-value LDS race, all addresses in-bounds; every
  // load/store below is unconditional straight-line code).
  int zrow[8], zcol[8];
  float zreg[8];
  #pragma unroll
  for (int i = 0; i < 8; ++i) {
    int idx = i * NT + tid;
    idx = (idx < ZELT) ? idx : (ZELT - 1);
    zrow[i] = idx / 63;
    zcol[i] = idx % 63;
  }
  // Prologue: fetch z for t = 0.
  #pragma unroll
  for (int i = 0; i < 8; ++i)
    zreg[i] = z[((size_t)(b0 + zrow[i]) * 128 + 64) * 63 + zcol[i]];

  __syncthreads();

  // ---------------- 64-step recurrence ----------------
  for (int t = 0; t < PSTEPS; ++t) {
    // 1. Commit staged z_t into activation cols 1..63 (bf16).
    #pragma unroll
    for (int i = 0; i < 8; ++i)
      sAct[zrow[i] * AS + 1 + zcol[i]] = (__bf16)zreg[i];
    __syncthreads();

    // 2. Prefetch z_{t+1} into registers; latency hides under the WMMAs below.
    if (t + 1 < PSTEPS) {
      #pragma unroll
      for (int i = 0; i < 8; ++i)
        zreg[i] = z[((size_t)(b0 + zrow[i]) * 128 + 64 + (t + 1)) * 63 + zcol[i]];
    }

    // 3. A fragments (16x32 bf16, k = 0..5): M = lane%16, kb = half*8;
    //    two contiguous 16B LDS reads per fragment.
    v16bf afrag[6];
    #pragma unroll
    for (int k = 0; k < 6; ++k) {
      const __bf16* base = sAct + (mtile * 16 + n) * AS + k * 32 + half * 8;
      v8bf lo = *(const v8bf*)(base);
      v8bf hi = *(const v8bf*)(base + 16);
      #pragma unroll
      for (int e = 0; e < 8; ++e) { afrag[k][e] = lo[e]; afrag[k][8 + e] = hi[e]; }
    }

    __bf16 hnew[2][8];
    #pragma unroll
    for (int p = 0; p < 2; ++p) {
      const int htile = hbase + 4 * p;
      v8f acc[4];
      #pragma unroll
      for (int gi = 0; gi < 4; ++gi) {
        #pragma unroll
        for (int e = 0; e < 8; ++e) acc[gi][e] = 0.0f;   // inline-0 C operand
      }
      // Accumulate over K = 192 (6 k-tiles), 4 gate tiles per patch.
      #pragma unroll
      for (int k = 0; k < 6; ++k) {
        #pragma unroll
        for (int gi = 0; gi < 4; ++gi) {
          const __bf16* wb =
              sW + (gi * 128 + htile * 16 + n) * WS + k * 32 + half * 8;
          v8bf blo = *(const v8bf*)(wb);
          v8bf bhi = *(const v8bf*)(wb + 16);
          v16bf bfrag;
          #pragma unroll
          for (int e = 0; e < 8; ++e) { bfrag[e] = blo[e]; bfrag[8 + e] = bhi[e]; }
          acc[gi] = __builtin_amdgcn_wmma_f32_16x16x32_bf16(
              false, afrag[k], false, bfrag, (short)0, acc[gi], false, false);
        }
      }
      // LSTM cell update (bias folded here); c stays in VGPRs.
      #pragma unroll
      for (int v = 0; v < 8; ++v) {
        float gi_ = acc[0][v] + biasr[p][0];
        float gf_ = acc[1][v] + biasr[p][1];
        float gg_ = acc[2][v] + biasr[p][2];
        float go_ = acc[3][v] + biasr[p][3];
        float ci  = sigmoidf_(gf_) * creg[p][v] + sigmoidf_(gi_) * tanhf(gg_);
        creg[p][v] = ci;
        hnew[p][v] = (__bf16)(sigmoidf_(go_) * tanhf(ci));
      }
    }

    __syncthreads();   // all waves finished reading old h from LDS
    #pragma unroll
    for (int p = 0; p < 2; ++p) {
      const int hcol = (hbase + 4 * p) * 16 + n;
      #pragma unroll
      for (int v = 0; v < 8; ++v) {
        const int m = v + 8 * half;
        sAct[(mtile * 16 + m) * AS + 64 + hcol] = hnew[p][v];
      }
    }
    __syncthreads();   // new h visible

    // 4. Parallel readout: 8 threads per row, 16 FMAs each + shuffle tree.
    {
      const int r    = tid >> 3;
      const int part = tid & 7;
      const __bf16* hr = sAct + r * AS + 64 + part * 16;
      const float*  wo = sWout + part * 16;
      float accv = 0.0f;
      #pragma unroll
      for (int j = 0; j < 16; ++j) accv += (float)hr[j] * wo[j];
      accv += __shfl_xor(accv, 1, 32);
      accv += __shfl_xor(accv, 2, 32);
      accv += __shfl_xor(accv, 4, 32);
      if (part == 0) {
        const float o = accv + boutv;
        out[(size_t)(b0 + r) * PSTEPS + t] = o;
        sAct[r * AS + 0] = (__bf16)o;    // feedback into input col 0
      }
    }
    // No barrier needed: next iteration's post-z-commit barrier orders the
    // col-0 write before the next GEMM; z-commit touches only cols 1..63.
  }
}

extern "C" void kernel_launch(void* const* d_in, const int* in_sizes, int n_in,
                              void* d_out, int out_size, void* d_ws, size_t ws_size,
                              hipStream_t stream) {
  const float* xp    = (const float*)d_in[0];
  const float* zp    = (const float*)d_in[1];
  const float* h0p   = (const float*)d_in[2];
  const float* c0p   = (const float*)d_in[3];
  const float* Wihp  = (const float*)d_in[4];
  const float* Whhp  = (const float*)d_in[5];
  const float* bihp  = (const float*)d_in[6];
  const float* bhhp  = (const float*)d_in[7];
  const float* Woutp = (const float*)d_in[8];
  const float* boutp = (const float*)d_in[9];
  float* outp = (float*)d_out;

  const size_t shmem = (size_t)512 * WS * sizeof(__bf16)   // weights (bf16)
                     + (size_t)BT * AS * sizeof(__bf16)    // activations
                     + 128 * sizeof(float);                // W_out
  // Opt in to >64KB dynamic LDS (gfx1250 WGP has 320KB).
  hipFuncSetAttribute(reinterpret_cast<const void*>(lstm_decoder_wmma),
                      hipFuncAttributeMaxDynamicSharedMemorySize, (int)shmem);

  dim3 grid(8192 / BT);   // 256 workgroups
  lstm_decoder_wmma<<<grid, NT, shmem, stream>>>(
      xp, zp, h0p, c0p, Wihp, Whhp, bihp, bhhp, Woutp, boutp, outp);
}
